// Model_55963423867287
// MI455X (gfx1250) — matrix operand order repaired and checked by reading, not compiled, not run on hardware
//
#include <hip/hip_runtime.h>
#include <math.h>

// Problem constants (from the reference)
#define B_   512
#define S_   128
#define DIN  256
#define HID  256
#define J_   257
#define JP   272          // J padded to 17*16 for WMMA
#define NT1  16           // HID/16
#define NT2  17           // JP/16
#define KT   8            // 256/32 k-steps per GEMM

typedef __bf16 bf16_t;
typedef __attribute__((ext_vector_type(16))) __bf16 v16bf;
typedef __attribute__((ext_vector_type(8)))  __bf16 v8bf;
typedef __attribute__((ext_vector_type(8)))  float  v8f;
typedef __attribute__((ext_vector_type(4)))  float  v4f;

// Workspace layout (bytes)
static constexpr size_t OFF_SCAL = 0;                       // 3 floats: p0, log(p0), 1/(1-p0)
static constexpr size_t OFF_W1P  = 256;                     // packed W1 frags: 8*16*32*16 bf16 = 131072 B
static constexpr size_t OFF_WJP  = OFF_W1P + 65536u * 2;    // packed Wj frags: 8*17*32*16 bf16 = 139264 B
static constexpr size_t OFF_BJP  = OFF_WJP + 69632u * 2;    // padded bias: 272 f32
static constexpr size_t OFF_H    = 271872;                  // h bf16 [65536][256] = 33554432 B

static __device__ __forceinline__ v16bf join8(v8bf lo, v8bf hi) {
  return __builtin_shufflevector(lo, hi, 0,1,2,3,4,5,6,7,8,9,10,11,12,13,14,15);
}

// Branchless tanh: avoids OCML's saveexec-heavy path. (e^{2x}-1)/(e^{2x}+1),
// clamped so e stays finite; accuracy far above the bf16 noise floor.
static __device__ __forceinline__ float fast_tanh(float x) {
  x = fminf(fmaxf(x, -15.0f), 15.0f);
  float e = __expf(2.0f * x);
  return __fdividef(e - 1.0f, e + 1.0f);
}

// ---------------------------------------------------------------------------
// Kernel 1: pack weights into WMMA B-fragment order (bf16) + null path / p0.
// B-fragment layout (16x16x32 bf16, wave32):
//   lanes 0..15 : N = lane,    e0..7 -> K 0..7,  e8..15 -> K 16..23
//   lanes 16..31: N = lane-16, e0..7 -> K 8..15, e8..15 -> K 24..31
// ---------------------------------------------------------------------------
__global__ __launch_bounds__(256) void prep_kernel(
    const float* __restrict__ hvn, const float* __restrict__ W1,
    const float* __restrict__ b1,  const float* __restrict__ Wj,
    const float* __restrict__ bj,  const float* __restrict__ Wp,
    const float* __restrict__ bp,
    float* __restrict__ scal, bf16_t* __restrict__ w1p,
    bf16_t* __restrict__ wjp, float* __restrict__ bjp)
{
  __shared__ float hn[HID];
  if (blockIdx.x == 0) {
    // h_null = tanh(hvn @ W1 + b1); p0 = sigmoid(h_null @ Wp + bp)
    const int n = threadIdx.x;
    float acc = b1[n];
    for (int k = 0; k < DIN; ++k) acc += hvn[k] * W1[k * HID + n];
    hn[n] = fast_tanh(acc);
    __syncthreads();
    if (n == 0) {
      float s = bp[0];
      for (int k = 0; k < HID; ++k) s += hn[k] * Wp[k];
      float praw = 1.0f / (1.0f + __expf(-s));
      // mean of a broadcast scalar is the scalar itself
      float p0 = (praw > 0.5f) ? 0.1f : praw;
      scal[0] = p0;
      scal[1] = __logf(p0);
      scal[2] = 1.0f / (1.0f - p0);
    }
    return;
  }
  const int idx0   = (blockIdx.x - 1) * 256 + threadIdx.x;
  const int stride = (gridDim.x - 1) * 256;
  // Pack W1 -> [kt][nt][lane][e]
  for (int i = idx0; i < KT * NT1 * 32 * 16; i += stride) {
    int e    = i & 15;
    int lane = (i >> 4) & 31;
    int nt   = (i >> 9) & 15;
    int kt   = i >> 13;
    int k = kt * 32 + ((lane >> 4) << 3) + (e & 7) + ((e >> 3) << 4);
    int n = nt * 16 + (lane & 15);
    w1p[i] = (bf16_t)W1[k * HID + n];
  }
  // Pack Wj (padded cols 257..271 -> 0) -> [kt][nt][lane][e]
  for (int i = idx0; i < KT * NT2 * 32 * 16; i += stride) {
    int e    = i & 15;
    int lane = (i >> 4) & 31;
    int t    = i >> 9;
    int nt   = t % NT2;
    int kt   = t / NT2;
    int k = kt * 32 + ((lane >> 4) << 3) + (e & 7) + ((e >> 3) << 4);
    int n = nt * 16 + (lane & 15);
    wjp[i] = (n < J_) ? (bf16_t)Wj[k * J_ + n] : (bf16_t)0.0f;
  }
  // Padded bias (pad lanes get -1e30 so exp() underflows to 0 in softmax)
  for (int i = idx0; i < JP; i += stride) bjp[i] = (i < J_) ? bj[i] : -1.0e30f;
}

// ---------------------------------------------------------------------------
// Kernel 2: h = tanh(hv @ W1 + b1), bf16 WMMA, h stored bf16 row-major in ws.
// Loop nest: kt outermost, 8 persistent accumulators (one per nt) per wave.
// Per k-step: 16 b128 B-loads issued together, then 8 back-to-back WMMAs ->
// progressive s_wait_loadcnt instead of wait-0 before every WMMA.
// A-fragment layout (16x16x32 bf16): lane<16 -> M=lane, e0..7=K0..7, e8..15=K16..23
//                                    lane>=16 -> M=lane-16, e0..7=K8..15, e8..15=K24..31
// Block: 256 thr = 8 waves, 64 rows. Wave w: M-tile w&3, N-tiles (w>>2)*8 .. +7.
// ---------------------------------------------------------------------------
__global__ __launch_bounds__(256) void gemm1_tanh_kernel(
    const float* __restrict__ hv, const bf16_t* __restrict__ w1p,
    const float* __restrict__ b1, bf16_t* __restrict__ hws)
{
  const int lane   = threadIdx.x & 31;
  const int w      = threadIdx.x >> 5;
  const int m0     = blockIdx.x * 64;
  const int mt     = w & 3;
  const int ntBase = (w >> 2) * 8;
  const int halfHi = lane >> 4;
  const int row    = m0 + mt * 16 + (lane & 15);
  const int koff   = halfHi ? 8 : 0;

  // Preload + convert all A fragments (full K in registers)
  v16bf a[KT];
#pragma unroll
  for (int kt = 0; kt < KT; ++kt) {
    const float* p = hv + (size_t)row * DIN + kt * 32 + koff;
    v4f x0 = *(const v4f*)(p);
    v4f x1 = *(const v4f*)(p + 4);
    v4f x2 = *(const v4f*)(p + 16);
    v4f x3 = *(const v4f*)(p + 20);
    v16bf t;
#pragma unroll
    for (int e = 0; e < 4; ++e) {
      t[e]      = (bf16_t)x0[e];
      t[e + 4]  = (bf16_t)x1[e];
      t[e + 8]  = (bf16_t)x2[e];
      t[e + 12] = (bf16_t)x3[e];
    }
    a[kt] = t;
  }

  v8f acc[8];
#pragma unroll
  for (int ni = 0; ni < 8; ++ni)
    acc[ni] = (v8f){0.f, 0.f, 0.f, 0.f, 0.f, 0.f, 0.f, 0.f};

#pragma unroll
  for (int kt = 0; kt < KT; ++kt) {
    v16bf bfr[8];
#pragma unroll
    for (int ni = 0; ni < 8; ++ni) {
      const int nt = ntBase + ni;
      const v8bf* bptr = (const v8bf*)(w1p + (((size_t)(kt * NT1 + nt) * 32 + lane) << 4));
      bfr[ni] = join8(bptr[0], bptr[1]);
    }
#pragma unroll
    for (int ni = 0; ni < 8; ++ni)
      acc[ni] = __builtin_amdgcn_wmma_f32_16x16x32_bf16(false, a[kt], false, bfr[ni],
                                                        (short)0, acc[ni], false, false);
  }

#pragma unroll
  for (int ni = 0; ni < 8; ++ni) {
    const int nt   = ntBase + ni;
    const int col  = nt * 16 + (lane & 15);
    const float bi = b1[col];
#pragma unroll
    for (int v = 0; v < 8; ++v) {
      // C layout: VGPR v -> M = v (lanes 0..15) / v+8 (lanes 16..31), N = lane&15
      const int m = m0 + mt * 16 + v + halfHi * 8;
      hws[(size_t)m * HID + col] = (bf16_t)fast_tanh(acc[ni][v] + bi);
    }
  }
}

// ---------------------------------------------------------------------------
// Kernel 3: logits = h @ Wj + bj (WMMA) -> softmax (LDS, wave-per-row) ->
// band gather + normalize by 1/(1-p0) -> write transition & transition_log
// (band twice each + diag(p0)/diag(log p0) blocks), fully coalesced.
// Same kt-outer structure with persistent accumulators; 17 N-tiles are split
// wave-uniformly (wave w gets nt = (w&3)+4*ni, 5 tiles when (w&3)==0 else 4),
// so EXEC stays all-ones around every WMMA.
// Block: 256 thr = 8 waves, 32 rows (one batch slice). 2048 blocks total.
// ---------------------------------------------------------------------------
__global__ __launch_bounds__(256) void gemm2_softmax_out_kernel(
    const bf16_t* __restrict__ hws, const bf16_t* __restrict__ wjp,
    const float* __restrict__ bjp,  const float* __restrict__ scal,
    float* __restrict__ T, float* __restrict__ TL)
{
  __shared__ float slog[32 * 288];   // padded row stride avoids bank conflicts
  __shared__ float srec[32];
  const int lane   = threadIdx.x & 31;
  const int w      = threadIdx.x >> 5;
  const int g0     = blockIdx.x * 32;        // global row (b*S + i)
  const int b      = g0 >> 7;
  const int i0     = g0 & 127;
  const int mt     = w >> 2;                 // 0..1
  const int ntLo   = w & 3;
  const int cnt    = (ntLo == 0) ? 5 : 4;    // wave-uniform raggedness over 17 tiles
  const int halfHi = lane >> 4;
  const int row    = g0 + mt * 16 + (lane & 15);
  const int koff   = halfHi ? 8 : 0;

  // A fragments straight from bf16 h (two b128 loads per fragment)
  v16bf a[KT];
#pragma unroll
  for (int kt = 0; kt < KT; ++kt) {
    const bf16_t* p = hws + (size_t)row * HID + kt * 32 + koff;
    a[kt] = join8(*(const v8bf*)p, *(const v8bf*)(p + 16));
  }

  v8f acc[5];
#pragma unroll
  for (int ni = 0; ni < 5; ++ni)
    acc[ni] = (v8f){0.f, 0.f, 0.f, 0.f, 0.f, 0.f, 0.f, 0.f};

#pragma unroll
  for (int kt = 0; kt < KT; ++kt) {
    v16bf bfr[5];
#pragma unroll
    for (int ni = 0; ni < 5; ++ni) {
      if (ni < cnt) {
        const int nt = ntLo + 4 * ni;
        const v8bf* bptr = (const v8bf*)(wjp + (((size_t)(kt * NT2 + nt) * 32 + lane) << 4));
        bfr[ni] = join8(bptr[0], bptr[1]);
      }
    }
#pragma unroll
    for (int ni = 0; ni < 5; ++ni) {
      if (ni < cnt)
        acc[ni] = __builtin_amdgcn_wmma_f32_16x16x32_bf16(false, a[kt], false, bfr[ni],
                                                          (short)0, acc[ni], false, false);
    }
  }

#pragma unroll
  for (int ni = 0; ni < 5; ++ni) {
    if (ni < cnt) {
      const int nt   = ntLo + 4 * ni;
      const int col  = nt * 16 + (lane & 15);
      const float bi = bjp[col];
#pragma unroll
      for (int v = 0; v < 8; ++v)
        slog[(mt * 16 + v + halfHi * 8) * 288 + col] = acc[ni][v] + bi;
    }
  }
  __syncthreads();

  // Softmax: wave w owns rows 4w..4w+3; lane-strided scan + shfl_xor reduce
  for (int rr = 0; rr < 4; ++rr) {
    const int r = w * 4 + rr;
    float mx = -3.0e38f;
#pragma unroll
    for (int jj = 0; jj < 9; ++jj) {
      int c = lane + jj * 32;
      if (c < JP) mx = fmaxf(mx, slog[r * 288 + c]);
    }
#pragma unroll
    for (int off = 16; off > 0; off >>= 1) mx = fmaxf(mx, __shfl_xor(mx, off, 32));
    float s = 0.f;
#pragma unroll
    for (int jj = 0; jj < 9; ++jj) {
      int c = lane + jj * 32;
      if (c < JP) {
        float e = __expf(slog[r * 288 + c] - mx);
        slog[r * 288 + c] = e;
        s += e;
      }
    }
#pragma unroll
    for (int off = 16; off > 0; off >>= 1) s += __shfl_xor(s, off, 32);
    if (lane == 0) srec[r] = 1.0f / s;
  }
  __syncthreads();

  // Outputs: 256 threads = one full 256-wide row per iteration (coalesced)
  const float p0  = scal[0];
  const float lp0 = scal[1];
  const float inv = scal[2];    // 1/(1-p0); p0 is a broadcast scalar in the ref
  const int t = threadIdx.x;
  for (int r = 0; r < 32; ++r) {
    const int ii = i0 + r;
    float bv, lv;
    if (t < 128) {
      // band[b,ii,t] = jwp[b,ii, 128 + t - ii] / (1 - p0); index in [1,255]
      const float pn = slog[r * 288 + (128 + t - ii)] * srec[r] * inv;
      bv = pn;
      lv = __logf(pn);
    } else {
      const bool d = (t - 128) == ii;      // diag(p0) / diag(log p0)
      bv = d ? p0 : 0.f;
      lv = d ? lp0 : 0.f;
    }
    const size_t base = (size_t)b * (256 * 256) + (size_t)ii * 256 + t;
    T[base]               = bv;            // transition row ii
    T[base + 128 * 256]   = bv;            // transition row ii+128 (top repeated)
    TL[base]              = lv;            // transition_log row ii
    TL[base + 128 * 256]  = lv;            // transition_log row ii+128
  }
}

// ---------------------------------------------------------------------------
extern "C" void kernel_launch(void* const* d_in, const int* in_sizes, int n_in,
                              void* d_out, int out_size, void* d_ws, size_t ws_size,
                              hipStream_t stream) {
  (void)in_sizes; (void)n_in; (void)out_size; (void)ws_size;
  const float* hv  = (const float*)d_in[0];  // [512,128,256]
  const float* hvn = (const float*)d_in[1];  // [1,256]
  const float* W1  = (const float*)d_in[2];  // [256,256]
  const float* b1  = (const float*)d_in[3];  // [256]
  const float* Wj  = (const float*)d_in[4];  // [256,257]
  const float* bj  = (const float*)d_in[5];  // [257]
  const float* Wp  = (const float*)d_in[6];  // [256,1]
  const float* bp  = (const float*)d_in[7];  // [1]

  char*   ws   = (char*)d_ws;
  float*  scal = (float*)(ws + OFF_SCAL);
  bf16_t* w1p  = (bf16_t*)(ws + OFF_W1P);
  bf16_t* wjp  = (bf16_t*)(ws + OFF_WJP);
  float*  bjp  = (float*)(ws + OFF_BJP);
  bf16_t* hws  = (bf16_t*)(ws + OFF_H);

  float* T  = (float*)d_out;
  float* TL = T + (size_t)B_ * 256 * 256;

  prep_kernel<<<531, 256, 0, stream>>>(hvn, W1, b1, Wj, bj, Wp, bp,
                                       scal, w1p, wjp, bjp);
  gemm1_tanh_kernel<<<(B_ * S_) / 64, 256, 0, stream>>>(hv, w1p, b1, hws);
  gemm2_softmax_out_kernel<<<(B_ * S_) / 32, 256, 0, stream>>>(hws, wjp, bjp,
                                                               scal, T, TL);
}
